// PointTransformer_10857677325129
// MI455X (gfx1250) — matrix operand order repaired and checked by reading, not compiled, not run on hardware
//
#include <hip/hip_runtime.h>
#include <hip/hip_bf16.h>
#include <math.h>

// Problem constants (match reference)
#define NPTS  8192
#define CIN   256
#define COUT  256
#define NSAMP 16
#define SHARE 8
#define HW    32     // OUT / S
#define EPSV  1e-5f

typedef __attribute__((ext_vector_type(16))) __bf16 v16bf;
typedef __attribute__((ext_vector_type(8)))  float  v8f;
typedef __attribute__((ext_vector_type(4)))  unsigned int u32x4;
typedef __attribute__((ext_vector_type(4)))  int  i32x4;
typedef __attribute__((ext_vector_type(8)))  int  i32x8;

union Frag16 { v16bf v; unsigned int u[8]; };

__device__ __forceinline__ unsigned short f2bf(float f) {
  unsigned int u = __float_as_uint(f);
  u += 0x7FFFu + ((u >> 16) & 1u);      // round-to-nearest-even
  return (unsigned short)(u >> 16);
}

__device__ __forceinline__ v8f wmma_bf16(const Frag16& a, const Frag16& b, v8f c) {
  // D = A(16x32 bf16) * B(32x16 bf16) + C(16x16 f32)
  return __builtin_amdgcn_wmma_f32_16x16x32_bf16(false, a.v, false, b.v,
                                                 (short)0, c, false, false);
}

// ---------------------------------------------------------------------------
// TDM: async DMA of a 2D f32 tile (tile_w x tile_h elements) from a strided
// 2D tensor in global memory into LDS.  D# per CDNA5 ISA Tensor DMA Descriptor.
// Issue once per wave (EXEC-ignored scalar-descriptor op); caller must
// s_wait_tensorcnt before consumers read the LDS tile.
// ---------------------------------------------------------------------------
#if __has_builtin(__builtin_amdgcn_tensor_load_to_lds)
#define HAVE_TDM 1
__device__ __forceinline__ void tdm_load_tile_f32(
    const float* gsrc, void* lds_dst,
    unsigned tensor_w, unsigned tensor_h, unsigned row_stride_elems,
    unsigned tile_w, unsigned tile_h)
{
  const unsigned long long ga = (unsigned long long)(const void*)gsrc;
  const unsigned ldsa = (unsigned)(unsigned long long)lds_dst; // LDS byte offset (low 32 bits of flat)
  u32x4 g0;
  g0.x = 1u;                                            // count=1, user mode, no gather
  g0.y = ldsa;                                          // lds_addr
  g0.z = (unsigned)(ga & 0xFFFFFFFFu);                  // global_addr[31:0]
  g0.w = (unsigned)((ga >> 32) & 0x1FFFFFFu) | (2u << 30); // addr[56:32] | type=2
  i32x8 g1;
  g1[0] = (int)(2u << 16);                              // wg_mask=0, data_size=2 (4B)
  g1[1] = (int)((tensor_w & 0xFFFFu) << 16);            // tensor_dim0[15:0] @ bits 63:48
  g1[2] = (int)((tensor_w >> 16) | ((tensor_h & 0xFFFFu) << 16)); // dim0 hi | dim1 lo
  g1[3] = (int)((tensor_h >> 16) | (tile_w << 16));     // dim1 hi | tile_dim0
  g1[4] = (int)(tile_h & 0xFFFFu);                      // tile_dim1 (tile_dim2=0)
  g1[5] = (int)row_stride_elems;                        // tensor_dim0_stride[31:0]
  g1[6] = 0;                                            // stride0 hi | stride1 lo
  g1[7] = 0;                                            // stride1 hi
  i32x4 gz = {0, 0, 0, 0};
#if defined(__clang_major__) && (__clang_major__ >= 23)
  i32x8 gz8 = {0, 0, 0, 0, 0, 0, 0, 0};
  __builtin_amdgcn_tensor_load_to_lds(g0, g1, gz, gz, gz8, 0);
#else
  __builtin_amdgcn_tensor_load_to_lds(g0, g1, gz, gz, 0);
#endif
}
#else
#define HAVE_TDM 0
#endif

// ---------------------------------------------------------------------------
// Kernel A: fused QKV projection GEMM.  out = x @ W + b for W in {w_q,w_k,w_v}
// block = 256 thr (8 waves); each block computes a 128x64 tile of one output.
// A-tile staged global->LDS by the Tensor Data Mover (async, TENSORcnt);
// bf16 conversion happens at fragment-build time.
// ---------------------------------------------------------------------------
__global__ __launch_bounds__(256) void qkv_gemm(
    const float* __restrict__ x,
    const float* __restrict__ wq, const float* __restrict__ bq,
    const float* __restrict__ wk, const float* __restrict__ bk,
    const float* __restrict__ wv, const float* __restrict__ bv,
    float* __restrict__ xq, float* __restrict__ xk, float* __restrict__ xv)
{
  __shared__ float          saf[128 * 32];  // A tile, [row][k] f32 (TDM target)
  __shared__ unsigned short sb[64 * 32];    // B tile, [col][k] bf16

  const int tid  = threadIdx.x;
  const int lane = tid & 31;
  const int wave = tid >> 5;
  const int m0   = blockIdx.x * 128;
  const int mat  = blockIdx.y >> 2;               // 0=q 1=k 2=v
  const int n0   = (blockIdx.y & 3) * 64;

  const float* W  = (mat == 0) ? wq : (mat == 1) ? wk : wv;
  const float* BS = (mat == 0) ? bq : (mat == 1) ? bk : bv;
  float*       O  = (mat == 0) ? xq : (mat == 1) ? xk : xv;

  v8f acc[4] = {};
  const int kmapA[8] = {0, 2, 4, 6, 16, 18, 20, 22};
  const int kbA = (lane >> 4) * 8;
  const int kbB = (lane >> 4) * 16;

  for (int k0 = 0; k0 < CIN; k0 += 32) {
#if HAVE_TDM
    if (wave == 0) {
      // 32x128 f32 tile of x starting at (m0, k0); row stride CIN elements
      tdm_load_tile_f32(x + (size_t)m0 * CIN + k0, (void*)saf,
                        /*tensor_w=*/CIN, /*tensor_h=*/NPTS,
                        /*row_stride=*/CIN, /*tile_w=*/32, /*tile_h=*/128);
      __builtin_amdgcn_s_wait_tensorcnt(0);
    }
#else
    for (int e = tid; e < 128 * 32; e += 256) {
      int r = e >> 5, kk = e & 31;
      saf[e] = x[(size_t)(m0 + r) * CIN + k0 + kk];
    }
#endif
    for (int e = tid; e < 64 * 32; e += 256) {
      int kk = e >> 6, nn = e & 63;
      sb[nn * 32 + kk] = f2bf(W[(size_t)(k0 + kk) * COUT + n0 + nn]);
    }
    __syncthreads();

    Frag16 af;
    const int arow = wave * 16 + (lane & 15);
#pragma unroll
    for (int j = 0; j < 8; j++) {
      const float2 f = *(const float2*)&saf[arow * 32 + kbA + kmapA[j]];
      af.u[j] = (unsigned int)f2bf(f.x) | ((unsigned int)f2bf(f.y) << 16);
    }

#pragma unroll
    for (int nt = 0; nt < 4; nt++) {
      Frag16 bf;
      const int bcol = nt * 16 + (lane & 15);
#pragma unroll
      for (int j = 0; j < 8; j++)
        bf.u[j] = *(const unsigned int*)&sb[bcol * 32 + kbA + kmapA[j]];
      acc[nt] = wmma_bf16(af, bf, acc[nt]);
    }
    __syncthreads();
  }

  const int hi = lane >> 4;
#pragma unroll
  for (int nt = 0; nt < 4; nt++) {
    const int col  = n0 + nt * 16 + (lane & 15);
    const float bb = BS[col];
#pragma unroll
    for (int e = 0; e < 8; e++) {
      const int row = m0 + wave * 16 + e + 8 * hi;
      O[(size_t)row * COUT + col] = acc[nt][e] + bb;
    }
  }
}

// ---------------------------------------------------------------------------
// Kernel B: brute-force KNN, top-16 per query. Candidates tiled through LDS.
// ---------------------------------------------------------------------------
__global__ __launch_bounds__(256) void knn_topk(
    const float* __restrict__ p, int* __restrict__ knn)
{
  __shared__ float sp[2048 * 3];
  const int q = blockIdx.x * 256 + threadIdx.x;
  const float qx = p[q * 3 + 0], qy = p[q * 3 + 1], qz = p[q * 3 + 2];

  float bd[NSAMP]; int bi[NSAMP];
#pragma unroll
  for (int t = 0; t < NSAMP; t++) { bd[t] = 3.4e38f; bi[t] = q; }
  float worst = 3.4e38f;

  for (int c0 = 0; c0 < NPTS; c0 += 2048) {
    __syncthreads();
    for (int e = threadIdx.x; e < 2048 * 3; e += 256) sp[e] = p[c0 * 3 + e];
    __syncthreads();
    for (int j = 0; j < 2048; j++) {
      const float dx = sp[j * 3 + 0] - qx;
      const float dy = sp[j * 3 + 1] - qy;
      const float dz = sp[j * 3 + 2] - qz;
      const float d = dx * dx + dy * dy + dz * dz;
      if (d < worst) {
        float wv = bd[0]; int wj = 0;
#pragma unroll
        for (int t = 1; t < NSAMP; t++) if (bd[t] > wv) { wv = bd[t]; wj = t; }
#pragma unroll
        for (int t = 0; t < NSAMP; t++) if (t == wj) { bd[t] = d; bi[t] = c0 + j; }
        worst = bd[0];
#pragma unroll
        for (int t = 1; t < NSAMP; t++) worst = fmaxf(worst, bd[t]);
      }
    }
  }
#pragma unroll
  for (int t = 0; t < NSAMP; t++) knn[q * NSAMP + t] = bi[t];
}

// ---------------------------------------------------------------------------
// Kernel C: fused per-point attention. One wave32 per point; 8 points / block.
// linear_p bottleneck -> r_qk -> WMMA linear_w (256->32, 32->32) -> shfl
// softmax over neighbors -> gathered weighted aggregation.
// ---------------------------------------------------------------------------
__global__ __launch_bounds__(256) void pt_attn(
    const float* __restrict__ p,
    const float* __restrict__ p_w1, const float* __restrict__ p_b1,
    const float* __restrict__ p_bn_g, const float* __restrict__ p_bn_b,
    const float* __restrict__ p_bn_m, const float* __restrict__ p_bn_v,
    const float* __restrict__ p_w2, const float* __restrict__ p_b2,
    const float* __restrict__ bn1_g, const float* __restrict__ bn1_b,
    const float* __restrict__ bn1_m, const float* __restrict__ bn1_v,
    const float* __restrict__ w_l1_w, const float* __restrict__ w_l1_b,
    const float* __restrict__ bn2_g, const float* __restrict__ bn2_b,
    const float* __restrict__ bn2_m, const float* __restrict__ bn2_v,
    const float* __restrict__ w_l2_w, const float* __restrict__ w_l2_b,
    const float* __restrict__ xq, const float* __restrict__ xk,
    const float* __restrict__ xv, const int* __restrict__ knn,
    float* __restrict__ out)
{
  __shared__ unsigned short sw1[COUT * HW];   // w_l1_w as bf16, [h][k]
  __shared__ unsigned short sw2[HW * HW];     // w_l2_w as bf16, [ho][k]
  __shared__ float spw2[3 * COUT];            // p_w2
  __shared__ float spb2[COUT];                // p_b2
  __shared__ float ss1[COUT], ssh1[COUT];     // folded BN1 scale/shift
  __shared__ float ss2[HW], ssh2[HW], sb1l[HW], sb2l[HW];
  __shared__ float sh[8][NSAMP * 3];          // per-wave linear_p bottleneck
  __shared__ int   sidx[8][NSAMP];            // per-wave neighbor idx
  __shared__ unsigned short sa2[8][NSAMP * HW]; // bf16 A tile for 2nd GEMM
  __shared__ float swout[8][NSAMP * HW];      // softmaxed weights

  const int tid  = threadIdx.x;
  const int lane = tid & 31;
  const int wave = tid >> 5;

  // ---- Phase 0: stage weights / folded BN params into LDS ----
  for (int e = tid; e < COUT * HW; e += 256) {
    int h = e >> 8, k = e & 255;
    sw1[e] = f2bf(w_l1_w[(size_t)k * HW + h]);
  }
  for (int e = tid; e < HW * HW; e += 256) {
    int ho = e >> 5, k = e & 31;
    sw2[e] = f2bf(w_l2_w[(size_t)k * HW + ho]);
  }
  for (int e = tid; e < 3 * COUT; e += 256) spw2[e] = p_w2[e];
  if (tid < COUT) {
    spb2[tid] = p_b2[tid];
    const float s = bn1_g[tid] * rsqrtf(bn1_v[tid] + EPSV);
    ss1[tid]  = s;
    ssh1[tid] = bn1_b[tid] - bn1_m[tid] * s;
  }
  if (tid < HW) {
    const float s = bn2_g[tid] * rsqrtf(bn2_v[tid] + EPSV);
    ss2[tid]  = s;
    ssh2[tid] = bn2_b[tid] - bn2_m[tid] * s;
    sb1l[tid] = w_l1_b[tid];
    sb2l[tid] = w_l2_b[tid];
  }
  __syncthreads();

  const int n = blockIdx.x * 8 + wave;

  // ---- Phase 1: neighbor indices + linear_p bottleneck h = relu(bn(dp@W1+b)) ----
  if (lane < NSAMP) {
    const int j = knn[n * NSAMP + lane];
    sidx[wave][lane] = j;
    const float dx = p[j * 3 + 0] - p[n * 3 + 0];
    const float dy = p[j * 3 + 1] - p[n * 3 + 1];
    const float dz = p[j * 3 + 2] - p[n * 3 + 2];
#pragma unroll
    for (int d = 0; d < 3; d++) {
      const float u = dx * p_w1[0 * 3 + d] + dy * p_w1[1 * 3 + d] +
                      dz * p_w1[2 * 3 + d] + p_b1[d];
      const float s = p_bn_g[d] * rsqrtf(p_bn_v[d] + EPSV);
      sh[wave][lane * 3 + d] = fmaxf((u - p_bn_m[d]) * s + p_bn_b[d], 0.f);
    }
  }
  __syncthreads();

  // ---- Phase 2: first linear_w GEMM, M=16(ns) K=256 N=32 via WMMA ----
  const int row  = lane & 15;     // neighbor t (A row)
  const int hi   = lane >> 4;
  const int kbA  = hi * 8;
  const int kbB  = hi * 16;
  const int colb = lane & 15;     // output column within tile
  const int kmapA[8] = {0, 2, 4, 6, 16, 18, 20, 22};

  const int jn = sidx[wave][row];
  const float* xkr = xk + (size_t)jn * COUT;
  const float* xqr = xq + (size_t)n  * COUT;
  const float h0 = sh[wave][row * 3 + 0];
  const float h1 = sh[wave][row * 3 + 1];
  const float h2 = sh[wave][row * 3 + 2];

  v8f acc1a = {}, acc1b = {};
  for (int k0 = 0; k0 < CIN; k0 += 32) {
    Frag16 af, bf0, bf1;
#pragma unroll
    for (int j = 0; j < 8; j++) {
      const int K = k0 + kbA + kmapA[j];
      unsigned int pk = 0;
#pragma unroll
      for (int e = 0; e < 2; e++) {
        const int Ke = K + e;
        const float pr = spb2[Ke] + h0 * spw2[Ke] + h1 * spw2[COUT + Ke] +
                         h2 * spw2[2 * COUT + Ke];
        float v = xkr[Ke] - xqr[Ke] + pr;          // r_qk
        v = fmaxf(v * ss1[Ke] + ssh1[Ke], 0.f);    // bn1 + relu
        pk |= (unsigned int)f2bf(v) << (16 * e);
      }
      af.u[j]  = pk;
      bf0.u[j] = *(const unsigned int*)&sw1[colb * COUT + k0 + kbA + kmapA[j]];
      bf1.u[j] = *(const unsigned int*)&sw1[(colb + 16) * COUT + k0 + kbA + kmapA[j]];
    }
    acc1a = wmma_bf16(af, bf0, acc1a);
    acc1b = wmma_bf16(af, bf1, acc1b);
  }

  // ---- Phase 3: bias + bn2 + relu, restage C-layout -> A-layout in LDS ----
#pragma unroll
  for (int e = 0; e < 8; e++) {
    const int rowm = e + 8 * hi;
    {
      float v = acc1a[e] + sb1l[colb];
      v = fmaxf(v * ss2[colb] + ssh2[colb], 0.f);
      sa2[wave][rowm * HW + colb] = f2bf(v);
    }
    {
      const int c2 = colb + 16;
      float v = acc1b[e] + sb1l[c2];
      v = fmaxf(v * ss2[c2] + ssh2[c2], 0.f);
      sa2[wave][rowm * HW + c2] = f2bf(v);
    }
  }
  __syncthreads();

  // ---- Phase 4: second linear_w GEMM (16x32 @ 32x32), one WMMA per tile ----
  Frag16 a2, b2a, b2b;
#pragma unroll
  for (int j = 0; j < 8; j++) {
    a2.u[j]  = *(const unsigned int*)&sa2[wave][row * HW + kbA + kmapA[j]];
    b2a.u[j] = *(const unsigned int*)&sw2[colb * HW + kbA + kmapA[j]];
    b2b.u[j] = *(const unsigned int*)&sw2[(colb + 16) * HW + kbA + kmapA[j]];
  }
  v8f zc = {};
  v8f acc2a = wmma_bf16(a2, b2a, zc);
  v8f acc2b = wmma_bf16(a2, b2b, zc);

  // ---- Phase 5: softmax over 16 neighbors (rows), per column; lane^16 pair ----
#pragma unroll
  for (int tile = 0; tile < 2; tile++) {
    const int col = colb + 16 * tile;
    float vals[8];
#pragma unroll
    for (int e = 0; e < 8; e++)
      vals[e] = (tile == 0 ? acc2a[e] : acc2b[e]) + sb2l[col];
    float m = vals[0];
#pragma unroll
    for (int e = 1; e < 8; e++) m = fmaxf(m, vals[e]);
    m = fmaxf(m, __shfl_xor(m, 16));
    float ssum = 0.f;
#pragma unroll
    for (int e = 0; e < 8; e++) { vals[e] = __expf(vals[e] - m); ssum += vals[e]; }
    ssum += __shfl_xor(ssum, 16);
    const float inv = 1.f / ssum;
#pragma unroll
    for (int e = 0; e < 8; e++)
      swout[wave][(e + 8 * hi) * HW + col] = vals[e] * inv;
  }
  __syncthreads();

  // ---- Phase 6: out[s,i] = sum_t w[t,i] * (xv_g[t, s*32+i] + pr[t, s*32+i]) ----
  const int i = lane;   // channel within H
#pragma unroll
  for (int t = 0; t < NSAMP; t++)
    __builtin_prefetch((const void*)(xv + (size_t)sidx[wave][t] * COUT + i), 0, 1);
#pragma unroll
  for (int s8 = 0; s8 < SHARE; s8++) {
    const int c = s8 * HW + i;
    const float pw0 = spw2[c], pw1v = spw2[COUT + c], pw2v = spw2[2 * COUT + c];
    const float pb = spb2[c];
    float acc = 0.f;
#pragma unroll
    for (int t = 0; t < NSAMP; t++) {
      const int j = sidx[wave][t];
      const float pr = pb + sh[wave][t * 3 + 0] * pw0 +
                       sh[wave][t * 3 + 1] * pw1v + sh[wave][t * 3 + 2] * pw2v;
      acc += swout[wave][t * HW + i] * (xv[(size_t)j * COUT + c] + pr);
    }
    out[(size_t)n * COUT + c] = acc;
  }
}

// ---------------------------------------------------------------------------
extern "C" void kernel_launch(void* const* d_in, const int* in_sizes, int n_in,
                              void* d_out, int out_size, void* d_ws, size_t ws_size,
                              hipStream_t stream) {
  const float* p      = (const float*)d_in[0];
  const float* x      = (const float*)d_in[1];
  const float* w_q    = (const float*)d_in[2];
  const float* b_q    = (const float*)d_in[3];
  const float* w_k    = (const float*)d_in[4];
  const float* b_k    = (const float*)d_in[5];
  const float* w_v    = (const float*)d_in[6];
  const float* b_v    = (const float*)d_in[7];
  const float* p_w1   = (const float*)d_in[8];
  const float* p_b1   = (const float*)d_in[9];
  const float* p_bn_g = (const float*)d_in[10];
  const float* p_bn_b = (const float*)d_in[11];
  const float* p_bn_m = (const float*)d_in[12];
  const float* p_bn_v = (const float*)d_in[13];
  const float* p_w2   = (const float*)d_in[14];
  const float* p_b2   = (const float*)d_in[15];
  const float* bn1_g  = (const float*)d_in[16];
  const float* bn1_b  = (const float*)d_in[17];
  const float* bn1_m  = (const float*)d_in[18];
  const float* bn1_v  = (const float*)d_in[19];
  const float* w_l1_w = (const float*)d_in[20];
  const float* w_l1_b = (const float*)d_in[21];
  const float* bn2_g  = (const float*)d_in[22];
  const float* bn2_b  = (const float*)d_in[23];
  const float* bn2_m  = (const float*)d_in[24];
  const float* bn2_v  = (const float*)d_in[25];
  const float* w_l2_w = (const float*)d_in[26];
  const float* w_l2_b = (const float*)d_in[27];

  float* xq  = (float*)d_ws;
  float* xk  = xq + (size_t)NPTS * COUT;
  float* xv  = xk + (size_t)NPTS * COUT;
  int*   knn = (int*)(xv + (size_t)NPTS * COUT);
  float* out = (float*)d_out;

  qkv_gemm<<<dim3(NPTS / 128, 12), 256, 0, stream>>>(
      x, w_q, b_q, w_k, b_k, w_v, b_v, xq, xk, xv);
  knn_topk<<<dim3(NPTS / 256), 256, 0, stream>>>(p, knn);
  pt_attn<<<dim3(NPTS / 8), 256, 0, stream>>>(
      p, p_w1, p_b1, p_bn_g, p_bn_b, p_bn_m, p_bn_v, p_w2, p_b2,
      bn1_g, bn1_b, bn1_m, bn1_v, w_l1_w, w_l1_b,
      bn2_g, bn2_b, bn2_m, bn2_v, w_l2_w, w_l2_b,
      xq, xk, xv, knn, out);
}
